// MultiHeadAttention_26852135534874
// MI455X (gfx1250) — matrix operand order repaired and checked by reading, not compiled, not run on hardware
//
#include <hip/hip_runtime.h>
#include <stdint.h>

// ---------------------------------------------------------------------------
// MI455X (gfx1250) fused multi-head attention, wave32 + WMMA f16->f32.
// Pipeline: cvt f32->f16 -> 3x proj GEMM (f16, async-LDS double buffered)
//           -> flash attention (f16 WMMA, f32 softmax) -> output GEMM (f32 out)
// ---------------------------------------------------------------------------

typedef __attribute__((ext_vector_type(16))) _Float16 v16h;
typedef __attribute__((ext_vector_type(8)))  _Float16 v8h;
typedef __attribute__((ext_vector_type(8)))  float    v8f;
typedef __attribute__((ext_vector_type(4)))  float    v4f;
typedef __attribute__((ext_vector_type(4)))  int      v4i;

union V16H { v16h v; v8h h[2]; };

#define D_MODEL   1024
#define NUM_HEADS 16
#define DEPTH     64
#define BATCH     4
#define SEQ       2048
#define M_TOTAL   (BATCH * SEQ)   // 8192

// ---------------------------------------------------------------------------
// CDNA5 async global->LDS staging (ASYNCcnt). Guarded so we always compile.
// Builtin signature (from clang diagnostic): (v4i as1*, v4i as3*, Imm, Imm)
// ---------------------------------------------------------------------------
#if defined(__gfx1250__) && __has_builtin(__builtin_amdgcn_global_load_async_to_lds_b128)
#define ASYNC_LDS 1
#else
#define ASYNC_LDS 0
#endif

#if ASYNC_LDS
using gv4i_p = __attribute__((address_space(1))) v4i*;  // global int4*
using lv4i_p = __attribute__((address_space(3))) v4i*;  // LDS int4*

__device__ __forceinline__ void async_ld_b128(const void* g, void* l) {
    // generic LDS address: low 32 bits are the LDS byte offset (aperture rule)
    __builtin_amdgcn_global_load_async_to_lds_b128(
        (gv4i_p)(uintptr_t)g,
        (lv4i_p)(uint32_t)(uintptr_t)l, 0, 0);
}

template<int N>
__device__ __forceinline__ void wait_async() {
#if __has_builtin(__builtin_amdgcn_s_wait_asynccnt)
    __builtin_amdgcn_s_wait_asynccnt(N);
#else
    if (N == 0) asm volatile("s_wait_asynccnt 0x0" ::: "memory");
    else        asm volatile("s_wait_asynccnt 0x4" ::: "memory");
#endif
}
#endif

// ---------------------------------------------------------------------------
// f32 -> f16 conversion, 8 elements/thread (element counts are multiples of 8)
// ---------------------------------------------------------------------------
__global__ __launch_bounds__(256)
void cvt_f32_f16(const float* __restrict__ in, _Float16* __restrict__ out, int n8)
{
    const int i = blockIdx.x * 256 + threadIdx.x;
    if (i >= n8) return;
    const v4f* p = (const v4f*)(in + (size_t)i * 8);
    const v4f a = p[0], b = p[1];
    v8h o;
    #pragma unroll
    for (int j = 0; j < 4; j++) { o[j] = (_Float16)a[j]; o[j + 4] = (_Float16)b[j]; }
    *(v8h*)(out + (size_t)i * 8) = o;
}

// ---------------------------------------------------------------------------
// GEMM: C[M=8192,N=1024] = A[M,1024](f16) x W[N,1024](f16)^T + bias
//   OUT_MODE=0 : f16 output scattered to [B,H,S,64], y=(acc+bias)*outScale
//   OUT_MODE=1 : f32 output row-major [M,N], y=acc+bias
// 256 threads (8 waves), tile 128x128, K-step 32, double-buffered LDS.
// Wave grid 4(M) x 2(N): each wave 32x64 = 2x4 WMMA tiles.
// ---------------------------------------------------------------------------
template<int OUT_MODE>
__global__ __launch_bounds__(256)
void gemm_f16_kernel(const _Float16* __restrict__ A, const _Float16* __restrict__ W,
                     const float* __restrict__ bias, void* __restrict__ Cv,
                     float outScale)
{
    __shared__ _Float16 As[2][128 * 32];   // [buf][row][k]
    __shared__ _Float16 Bs[2][128 * 32];   // [buf][n][k]

    const int wgM  = blockIdx.y * 128;
    const int wgN  = blockIdx.x * 128;
    const int tid  = threadIdx.x;
    const int lane = tid & 31;
    const int wave = tid >> 5;
    const int waveM = (wave & 3) * 32;
    const int waveN = (wave >> 2) * 64;

    const v8f zero = {};
    v8f acc[2][4];
    for (int i = 0; i < 2; i++)
        for (int j = 0; j < 4; j++) acc[i][j] = zero;

    // staging addresses: each thread moves 2x16B for A and 2x16B for B per tile
    const int ldRow  = tid >> 1;          // 0..127
    const int ldK    = (tid & 1) * 16;    // 0 or 16 (halves)
    const int ldsOff = ldRow * 32 + ldK;
    const _Float16* Ag = A + (size_t)(wgM + ldRow) * D_MODEL + ldK;
    const _Float16* Wg = W + (size_t)(wgN + ldRow) * D_MODEL + ldK;

#if ASYNC_LDS
    auto issue_tile = [&](int buf, int kk) {
        async_ld_b128(Ag + kk,     &As[buf][ldsOff]);
        async_ld_b128(Ag + kk + 8, &As[buf][ldsOff + 8]);
        async_ld_b128(Wg + kk,     &Bs[buf][ldsOff]);
        async_ld_b128(Wg + kk + 8, &Bs[buf][ldsOff + 8]);
    };
    issue_tile(0, 0);
#endif

    const int arow = lane & 15;
    const int ak0  = (lane >> 4) * 8;
    const int bk0  = (lane >> 4) * 8;

    #pragma unroll 1
    for (int kt = 0; kt < D_MODEL / 32; kt++) {
        const int kk  = kt * 32;
        const int cur = kt & 1;

#if ASYNC_LDS
        if (kt + 1 < D_MODEL / 32) {
            issue_tile(cur ^ 1, kk + 32);   // prefetch next tile into other buf
            wait_async<4>();                // current tile's 4 ops complete
        } else {
            wait_async<0>();
        }
        __syncthreads();
#else
        {   // sync staging fallback (ping-pong still valid with 2 barriers)
            v8h a0 = *(const v8h*)(Ag + kk);
            v8h a1 = *(const v8h*)(Ag + kk + 8);
            v8h b0 = *(const v8h*)(Wg + kk);
            v8h b1 = *(const v8h*)(Wg + kk + 8);
            *(v8h*)(&As[cur][ldsOff])     = a0;
            *(v8h*)(&As[cur][ldsOff + 8]) = a1;
            *(v8h*)(&Bs[cur][ldsOff])     = b0;
            *(v8h*)(&Bs[cur][ldsOff + 8]) = b1;
        }
        __syncthreads();
#endif

        // ---- A fragments: 16x32 f16 per ISA layout ----
        V16H a[2];
        #pragma unroll
        for (int mt = 0; mt < 2; mt++) {
            const int r = waveM + mt * 16 + arow;
            a[mt].h[0] = *(const v8h*)(&As[cur][r * 32 + ak0]);
            a[mt].h[1] = *(const v8h*)(&As[cur][r * 32 + ak0 + 16]);
        }
        // ---- B fragments: k per ISA layout (8h..8h+7, 16+8h..) of one column ----
        #pragma unroll
        for (int nt = 0; nt < 4; nt++) {
            const int n = waveN + nt * 16 + (lane & 15);
            V16H bf;
            bf.h[0] = *(const v8h*)(&Bs[cur][n * 32 + bk0]);
            bf.h[1] = *(const v8h*)(&Bs[cur][n * 32 + bk0 + 16]);
            #pragma unroll
            for (int mt = 0; mt < 2; mt++) {
                acc[mt][nt] = __builtin_amdgcn_wmma_f32_16x16x32_f16(
                    false, a[mt].v, false, bf.v, (short)0, acc[mt][nt],
                    false, false);
            }
        }
        __syncthreads();
    }

    // ---- epilogue: C layout lane->(row=g+8*(lane>=16), col=lane&15) ----
    const int crow = (lane >> 4) * 8;
    const int ccol = lane & 15;
    #pragma unroll
    for (int mt = 0; mt < 2; mt++) {
        #pragma unroll
        for (int nt = 0; nt < 4; nt++) {
            const int nglob = wgN + waveN + nt * 16 + ccol;
            const float bv  = bias[nglob];
            #pragma unroll
            for (int g = 0; g < 8; g++) {
                const int m = wgM + waveM + mt * 16 + crow + g;
                const float y = (acc[mt][nt][g] + bv) * outScale;
                if constexpr (OUT_MODE == 0) {
                    _Float16* out = (_Float16*)Cv;
                    const int b = m >> 11, s = m & (SEQ - 1);
                    const int h = nglob >> 6, d = nglob & (DEPTH - 1);
                    out[(((size_t)(b * NUM_HEADS + h) * SEQ) + s) * DEPTH + d] =
                        (_Float16)y;
                } else {
                    float* out = (float*)Cv;
                    out[(size_t)m * D_MODEL + nglob] = y;
                }
            }
        }
    }
}

// ---------------------------------------------------------------------------
// Flash attention: grid = B*H*(S/128), 256 threads (8 waves).
// Each wave owns 16 q-rows; 32-key blocks with online softmax.
// Q was pre-scaled by 1/sqrt(DEPTH)=0.125 in the projection epilogue.
// ---------------------------------------------------------------------------
__global__ __launch_bounds__(256)
void attn_kernel(const _Float16* __restrict__ Q, const _Float16* __restrict__ K,
                 const _Float16* __restrict__ V, const float* __restrict__ mask,
                 _Float16* __restrict__ O)
{
    __shared__ _Float16 Vt[DEPTH * 32];      // [d][key_local] transposed V block
    __shared__ _Float16 Ps[8 * 16 * 32];     // per-wave P tile [16 q][32 k]
    __shared__ float    Ms[8 * 16 * 32];     // per-wave mask tile [16 q][32 k]

    const int tid  = threadIdx.x;
    const int lane = tid & 31;
    const int wave = tid >> 5;

    const int qb = blockIdx.x & 15;          // q-block within sequence
    const int bh = blockIdx.x >> 4;          // b*16+h
    const int b  = bh >> 4;
    const int h  = bh & 15;

    const _Float16* Qbh = Q + (size_t)bh * SEQ * DEPTH;
    const _Float16* Kbh = K + (size_t)bh * SEQ * DEPTH;
    const _Float16* Vbh = V + (size_t)bh * SEQ * DEPTH;

    const int qrow0 = qb * 128 + wave * 16;

    // Q A-fragments for d-chunks [0,32) and [32,64)
    V16H qa[2];
    {
        const int r  = qrow0 + (lane & 15);
        const int k0 = (lane >> 4) * 8;
        const _Float16* qp = Qbh + (size_t)r * DEPTH;
        #pragma unroll
        for (int c = 0; c < 2; c++) {
            qa[c].h[0] = *(const v8h*)(qp + c * 32 + k0);
            qa[c].h[1] = *(const v8h*)(qp + c * 32 + k0 + 16);
        }
    }

    const v8f zero = {};
    v8f o[4];
    #pragma unroll
    for (int d = 0; d < 4; d++) o[d] = zero;
    float mrun[8], lrun[8];
    #pragma unroll
    for (int g = 0; g < 8; g++) { mrun[g] = -1e30f; lrun[g] = 0.0f; }

    const int crow = (lane >> 4) * 8;
    const int ccol = lane & 15;
    const size_t mbase = (size_t)b * SEQ * SEQ;   // mask is [B,1,S,S]

    // per-wave mask staging addresses (4x b128 per lane per iteration)
    float* Mw = &Ms[wave * 512];
    const int mr  = lane & 15;
    const int mc0 = (lane >> 4) * 16;

    #pragma unroll 1
    for (int kb = 0; kb < SEQ / 32; kb++) {
        const int kbase = kb * 32;

        __syncthreads();
        // cooperative transpose: V[32 keys][64 d] -> Vt[64 d][32 keys]
        {
            const int keyl = tid >> 3;           // 0..31
            const int d0   = (tid & 7) * 8;      // 0..56
            v8h tmp = *(const v8h*)(Vbh + (size_t)(kbase + keyl) * DEPTH + d0);
            #pragma unroll
            for (int i = 0; i < 8; i++) Vt[(d0 + i) * 32 + keyl] = tmp[i];
            if (kbase + 32 < SEQ)
                __builtin_prefetch(Vbh + (size_t)(kbase + 32 + keyl) * DEPTH + d0, 0, 1);
        }
        // per-wave mask tile: 16 rows x 32 cols f32 (coalesced 64B rows)
        {
            const float* gm = mask + mbase + (size_t)(qrow0 + mr) * SEQ + kbase + mc0;
            #pragma unroll
            for (int j = 0; j < 4; j++)
                *(v4f*)(&Mw[mr * 32 + mc0 + j * 4]) = ((const v4f*)gm)[j];
        }
        __syncthreads();

        // ---- scores S = Q K^T (B-fragment = d of one key, ISA layout) ----
        v8f s[2];
        #pragma unroll
        for (int nt = 0; nt < 2; nt++) {
            s[nt] = zero;
            const int key = kbase + nt * 16 + (lane & 15);
            const _Float16* kp = Kbh + (size_t)key * DEPTH + ((lane >> 4) * 8);
            #pragma unroll
            for (int c = 0; c < 2; c++) {
                V16H bf;
                bf.h[0] = *(const v8h*)(kp + c * 32);
                bf.h[1] = *(const v8h*)(kp + c * 32 + 16);
                s[nt] = __builtin_amdgcn_wmma_f32_16x16x32_f16(
                    false, qa[c].v, false, bf.v, (short)0, s[nt], false, false);
            }
        }

        // ---- mask + online softmax (rows live across 16-lane groups) ----
        float p0v[8], p1v[8];
        #pragma unroll
        for (int g = 0; g < 8; g++) {
            const int ro = (crow + g) * 32;
            float s0 = s[0][g] - 1e6f * Mw[ro + ccol];
            float s1 = s[1][g] - 1e6f * Mw[ro + 16 + ccol];
            float t = fmaxf(s0, s1);
            #pragma unroll
            for (int off = 8; off >= 1; off >>= 1)
                t = fmaxf(t, __shfl_xor(t, off, 32));
            const float mnew = fmaxf(mrun[g], t);
            const float corr = __expf(mrun[g] - mnew);
            const float p0 = __expf(s0 - mnew);
            const float p1 = __expf(s1 - mnew);
            float r = p0 + p1;
            #pragma unroll
            for (int off = 8; off >= 1; off >>= 1)
                r += __shfl_xor(r, off, 32);
            lrun[g] = lrun[g] * corr + r;
            mrun[g] = mnew;
            p0v[g] = p0; p1v[g] = p1;
            #pragma unroll
            for (int d = 0; d < 4; d++) o[d][g] *= corr;   // rescale O rows
        }

        // ---- P -> per-wave LDS (C layout -> A layout reshape) ----
        _Float16* Pw = &Ps[wave * 16 * 32];
        #pragma unroll
        for (int g = 0; g < 8; g++) {
            Pw[(crow + g) * 32 + ccol]      = (_Float16)p0v[g];
            Pw[(crow + g) * 32 + 16 + ccol] = (_Float16)p1v[g];
        }
        V16H pa;
        {
            const int pr  = lane & 15;
            const int pk0 = (lane >> 4) * 8;
            pa.h[0] = *(const v8h*)(&Pw[pr * 32 + pk0]);
            pa.h[1] = *(const v8h*)(&Pw[pr * 32 + pk0 + 16]);
        }

        // ---- O += P * V  (B-fragment from transposed Vt) ----
        const int bk0 = (lane >> 4) * 8;
        #pragma unroll
        for (int dt = 0; dt < 4; dt++) {
            const int dc = dt * 16 + (lane & 15);
            V16H bf;
            bf.h[0] = *(const v8h*)(&Vt[dc * 32 + bk0]);
            bf.h[1] = *(const v8h*)(&Vt[dc * 32 + bk0 + 16]);
            o[dt] = __builtin_amdgcn_wmma_f32_16x16x32_f16(
                false, pa.v, false, bf.v, (short)0, o[dt], false, false);
        }
    }

    // ---- normalize and store to [B,S,H*DEPTH] f16 for the Wo GEMM ----
    #pragma unroll
    for (int dt = 0; dt < 4; dt++) {
        #pragma unroll
        for (int g = 0; g < 8; g++) {
            const int qrow = qrow0 + crow + g;
            const int d    = dt * 16 + ccol;
            const float y  = o[dt][g] / lrun[g];
            O[((size_t)(b * SEQ + qrow) * D_MODEL) + h * DEPTH + d] = (_Float16)y;
        }
    }
}

// ---------------------------------------------------------------------------
extern "C" void kernel_launch(void* const* d_in, const int* in_sizes, int n_in,
                              void* d_out, int out_size, void* d_ws, size_t ws_size,
                              hipStream_t stream)
{
    (void)in_sizes; (void)n_in; (void)out_size; (void)ws_size;

    const float* q_in = (const float*)d_in[0];
    const float* k_in = (const float*)d_in[1];
    const float* v_in = (const float*)d_in[2];
    const float* m_in = (const float*)d_in[3];
    const float* Wq   = (const float*)d_in[4];
    const float* bq   = (const float*)d_in[5];
    const float* Wk   = (const float*)d_in[6];
    const float* bk   = (const float*)d_in[7];
    const float* Wv   = (const float*)d_in[8];
    const float* bv   = (const float*)d_in[9];
    const float* Wo   = (const float*)d_in[10];
    const float* bo   = (const float*)d_in[11];

    // Workspace layout (f16 buffers)
    char* ws = (char*)d_ws;
    const size_t XSZ = (size_t)M_TOTAL * D_MODEL * sizeof(_Float16);  // 16 MiB
    const size_t WSZ = (size_t)D_MODEL * D_MODEL * sizeof(_Float16);  //  2 MiB
    _Float16* Xq  = (_Float16*)(ws);                       // f16 inputs
    _Float16* Xk  = (_Float16*)(ws + 1 * XSZ);
    _Float16* Xv  = (_Float16*)(ws + 2 * XSZ);
    _Float16* Whq = (_Float16*)(ws + 3 * XSZ);             // f16 weights
    _Float16* Whk = (_Float16*)(ws + 3 * XSZ + 1 * WSZ);
    _Float16* Whv = (_Float16*)(ws + 3 * XSZ + 2 * WSZ);
    _Float16* Who = (_Float16*)(ws + 3 * XSZ + 3 * WSZ);
    _Float16* Qb  = (_Float16*)(ws + 3 * XSZ + 4 * WSZ);   // [B,H,S,64]
    _Float16* Kb  = (_Float16*)(ws + 4 * XSZ + 4 * WSZ);
    _Float16* Vb  = (_Float16*)(ws + 5 * XSZ + 4 * WSZ);
    _Float16* Ob  = (_Float16*)(ws + 6 * XSZ + 4 * WSZ);   // [B,S,1024]

    const int nX8 = M_TOTAL * D_MODEL / 8;    // 1048576
    const int nW8 = D_MODEL * D_MODEL / 8;    //  131072
    cvt_f32_f16<<<nX8 / 256, 256, 0, stream>>>(q_in, Xq, nX8);
    cvt_f32_f16<<<nX8 / 256, 256, 0, stream>>>(k_in, Xk, nX8);
    cvt_f32_f16<<<nX8 / 256, 256, 0, stream>>>(v_in, Xv, nX8);
    cvt_f32_f16<<<nW8 / 256, 256, 0, stream>>>(Wq, Whq, nW8);
    cvt_f32_f16<<<nW8 / 256, 256, 0, stream>>>(Wk, Whk, nW8);
    cvt_f32_f16<<<nW8 / 256, 256, 0, stream>>>(Wv, Whv, nW8);
    cvt_f32_f16<<<nW8 / 256, 256, 0, stream>>>(Wo, Who, nW8);

    dim3 ggrid(D_MODEL / 128, M_TOTAL / 128);   // (8, 64)

    // Projections (Q pre-scaled by 1/sqrt(DEPTH))
    gemm_f16_kernel<0><<<ggrid, 256, 0, stream>>>(Xq, Whq, bq, Qb, 0.125f);
    gemm_f16_kernel<0><<<ggrid, 256, 0, stream>>>(Xk, Whk, bk, Kb, 1.0f);
    gemm_f16_kernel<0><<<ggrid, 256, 0, stream>>>(Xv, Whv, bv, Vb, 1.0f);

    // Attention: B*H * (S/128) = 1024 workgroups
    attn_kernel<<<BATCH * NUM_HEADS * (SEQ / 128), 256, 0, stream>>>(
        Qb, Kb, Vb, m_in, Ob);

    // Output projection -> f32 d_out
    gemm_f16_kernel<1><<<ggrid, 256, 0, stream>>>(Ob, Who, bo, d_out, 1.0f);
}
